// Encoder_15522011808370
// MI455X (gfx1250) — matrix operand order repaired and checked by reading, not compiled, not run on hardware
//
#include <hip/hip_runtime.h>
#include <math.h>

typedef __attribute__((ext_vector_type(16))) __bf16 v16bf;
typedef __attribute__((ext_vector_type(8)))  float  v8f;

#define NN      50000
#define NE      800000
#define SPAN    768
#define EMB     128
#define KDIM    896
#define HID     256
#define NTYPES  8
#define KT_TOTAL 28   // 896 / 32
#define NT_TOTAL 16   // 256 / 16
#define BM      32    // rows per block

// ---------------------------------------------------------------------------
// Prepare W fragments: bf16 hi/lo split, fragment-major layout matching the
// contiguous k order (k = 16*h + e) that gemm_xw's A fragments also use:
//   lane 0..15  -> column n = lane,     K = e      (e = 0..15)
//   lane 16..31 -> column n = lane-16,  K = 16 + e
// Linear index: fidx = ((nt*28 + kt)*32 + lane)*16 + e
// ---------------------------------------------------------------------------
__global__ __launch_bounds__(256) void prep_w_frags(const float* __restrict__ W,
                                                    unsigned short* __restrict__ WhiU,
                                                    unsigned short* __restrict__ WloU)
{
    int idx = blockIdx.x * blockDim.x + threadIdx.x;
    if (idx >= NT_TOTAL * KT_TOTAL * 32 * 16) return;
    __bf16* Whi = reinterpret_cast<__bf16*>(WhiU);
    __bf16* Wlo = reinterpret_cast<__bf16*>(WloU);

    int e    = idx & 15;
    int lane = (idx >> 4) & 31;
    int rest = idx >> 9;
    int kt   = rest % KT_TOTAL;
    int nt   = rest / KT_TOTAL;

    int k = kt * 32 + ((lane >> 4) << 4) + e;
    int n = nt * 16 + (lane & 15);
    float w = W[(size_t)k * HID + n];
    __bf16 hi = (__bf16)w;
    __bf16 lo = (__bf16)(w - (float)hi);
    Whi[idx] = hi;
    Wlo[idx] = lo;
}

// ---------------------------------------------------------------------------
// GEMM: H[50000,256] = concat(span, emb[type]) @ W, bf16x3 split WMMA.
// Block = 256 threads = 8 waves; tile = 32 rows x 256 cols.
// wave m-index = wave>>2 (2 m-tiles), wave n-index = wave&3 (each 4 n-tiles).
// ---------------------------------------------------------------------------
__global__ __launch_bounds__(256) void gemm_xw(const float* __restrict__ span,
                                               const int*   __restrict__ types,
                                               const float* __restrict__ emb,
                                               const unsigned short* __restrict__ WhiU,
                                               const unsigned short* __restrict__ WloU,
                                               float* __restrict__ h)
{
    __shared__ float As[BM][36];          // stride 36 floats: 16B aligned rows, bank-spread
    __shared__ float embS[NTYPES * EMB];  // 4 KB: whole embedding table
    __shared__ int   typeS[BM];

    const __bf16* Whi = reinterpret_cast<const __bf16*>(WhiU);
    const __bf16* Wlo = reinterpret_cast<const __bf16*>(WloU);

    const int t    = threadIdx.x;
    const int lane = t & 31;
    const int wave = t >> 5;
    const int wm   = wave >> 2;      // 0..1
    const int wn   = wave & 3;       // 0..3
    const int block_row = blockIdx.x * BM;

    for (int i = t; i < NTYPES * EMB; i += 256) embS[i] = emb[i];
    if (t < BM) {
        int r = block_row + t;
        typeS[t] = (r < NN) ? types[r] : 0;
    }
    __syncthreads();

    v8f acc[4] = {};                 // 4 n-tiles of 16x16 f32

    const int hf   = lane >> 4;      // which K-half this lane holds
    const int m    = lane & 15;
    const int rowL = wm * 16 + m;    // local A row for this lane's fragment

    for (int kt = 0; kt < KT_TOTAL; ++kt) {
        // ---- stage A tile (32 rows x 32 k) into LDS, f32 ----
        {
            int r  = t >> 3;                 // 0..31
            int c0 = (t & 7) * 4;            // 0..28
            int gr = block_row + r;
            float4 v;
            if (gr < NN) {
                int k0 = kt * 32 + c0;
                if (kt < 24) {               // span_reps region (768 = 24*32)
                    v = *reinterpret_cast<const float4*>(span + (size_t)gr * SPAN + k0);
                } else {                     // embedding region (from LDS table)
                    const float* pe = embS + typeS[r] * EMB + (k0 - SPAN);
                    v = make_float4(pe[0], pe[1], pe[2], pe[3]);
                }
            } else {
                v = make_float4(0.f, 0.f, 0.f, 0.f);
            }
            *reinterpret_cast<float4*>(&As[r][c0]) = v;
        }
        __syncthreads();

        // ---- build A fragment (16-bit A 16x32 layout), f32 -> bf16 hi/lo ----
        v16bf ahi, alo;
        {
            const float* pa = &As[rowL][hf * 16];
            #pragma unroll
            for (int e = 0; e < 8; ++e) {
                float a0 = pa[e];            // K = hf*16 + e
                float a1 = pa[8 + e];        // K = hf*16 + 8 + e
                __bf16 h0 = (__bf16)a0;
                __bf16 h1 = (__bf16)a1;
                ahi[e]     = h0;
                ahi[8 + e] = h1;
                alo[e]     = (__bf16)(a0 - (float)h0);
                alo[8 + e] = (__bf16)(a1 - (float)h1);
            }
        }

        // ---- 4 n-tiles, 3 WMMAs each (hi*hi + hi*lo + lo*hi) ----
        #pragma unroll
        for (int nt = 0; nt < 4; ++nt) {
            int ntg = wn * 4 + nt;
            size_t base = ((size_t)(ntg * KT_TOTAL + kt) * 32 + lane) * 16;
            v16bf bhi = *reinterpret_cast<const v16bf*>(Whi + base);
            v16bf blo = *reinterpret_cast<const v16bf*>(Wlo + base);
            acc[nt] = __builtin_amdgcn_wmma_f32_16x16x32_bf16(
                false, ahi, false, bhi, (short)0, acc[nt], false, false);
            acc[nt] = __builtin_amdgcn_wmma_f32_16x16x32_bf16(
                false, ahi, false, blo, (short)0, acc[nt], false, false);
            acc[nt] = __builtin_amdgcn_wmma_f32_16x16x32_bf16(
                false, alo, false, bhi, (short)0, acc[nt], false, false);
        }
        __syncthreads();
    }

    // ---- store D (32-bit C/D 16x16 layout: lanes 0-15 M=r, 16-31 M=8+r) ----
    const int colBase = wn * 64 + (lane & 15);
    const int rowBase = block_row + wm * 16 + hf * 8;
    #pragma unroll
    for (int nt = 0; nt < 4; ++nt) {
        #pragma unroll
        for (int r = 0; r < 8; ++r) {
            int gr = rowBase + r;
            if (gr < NN) h[(size_t)gr * HID + colBase + nt * 16] = acc[nt][r];
        }
    }
}

// ---------------------------------------------------------------------------
// Graph propagation kernels
// ---------------------------------------------------------------------------
__global__ void deg_init(unsigned* __restrict__ deg)
{
    int i = blockIdx.x * blockDim.x + threadIdx.x;
    if (i < NN) deg[i] = 1u;                  // self-loop
}

__global__ void deg_count(const int* __restrict__ ei, unsigned* __restrict__ deg)
{
    int e = blockIdx.x * blockDim.x + threadIdx.x;
    if (e < NE) atomicAdd(&deg[ei[NE + e]], 1u);   // dst row of edge_index
}

__global__ void calc_dinv(const unsigned* __restrict__ deg, float* __restrict__ dinv)
{
    int i = blockIdx.x * blockDim.x + threadIdx.x;
    if (i < NN) dinv[i] = rsqrtf((float)deg[i]);   // deg >= 1 always
}

// out = h * dinv^2 (self-loop contribution); fully initializes d_out
__global__ void self_loop_init(const float* __restrict__ h,
                               const float* __restrict__ dinv,
                               float* __restrict__ out)
{
    int i = blockIdx.x * blockDim.x + threadIdx.x;
    if (i < NN * HID) {
        int n = i >> 8;
        float d = dinv[n];
        out[i] = h[i] * d * d;
    }
}

// wave-per-edge scatter: out[dst] += h[src] * dinv[src]*dinv[dst]
__global__ __launch_bounds__(256) void scatter_edges(const int*  __restrict__ ei,
                                                     const float* __restrict__ h,
                                                     const float* __restrict__ dinv,
                                                     float* __restrict__ out)
{
    int wid  = (blockIdx.x * blockDim.x + threadIdx.x) >> 5;
    int lane = threadIdx.x & 31;
    if (wid >= NE) return;
    int s = ei[wid];
    int d = ei[NE + wid];
    float nrm = dinv[s] * dinv[d];
    const float4* hs = reinterpret_cast<const float4*>(h + (size_t)s * HID);
    float* od = out + (size_t)d * HID;
    #pragma unroll
    for (int j = 0; j < 2; ++j) {
        int q = lane + j * 32;               // 64 float4 = 256 channels
        float4 v = hs[q];
        int c = q * 4;
        atomicAdd(&od[c + 0], v.x * nrm);
        atomicAdd(&od[c + 1], v.y * nrm);
        atomicAdd(&od[c + 2], v.z * nrm);
        atomicAdd(&od[c + 3], v.w * nrm);
    }
}

__global__ void finalize(const float* __restrict__ b,
                         const float* __restrict__ alpha,
                         float* __restrict__ out)
{
    int i = blockIdx.x * blockDim.x + threadIdx.x;
    if (i < NN * HID) {
        int c = i & 255;
        float v = out[i] + b[c];
        out[i] = (v > 0.f) ? v : alpha[c] * v;
    }
}

// ---------------------------------------------------------------------------
extern "C" void kernel_launch(void* const* d_in, const int* in_sizes, int n_in,
                              void* d_out, int out_size, void* d_ws, size_t ws_size,
                              hipStream_t stream)
{
    const float* span  = (const float*)d_in[0];
    const int*   types = (const int*)  d_in[1];
    const int*   ei    = (const int*)  d_in[2];
    const float* emb   = (const float*)d_in[3];
    const float* W     = (const float*)d_in[4];
    const float* b     = (const float*)d_in[5];
    const float* alpha = (const float*)d_in[6];
    float* out = (float*)d_out;

    char* ws = (char*)d_ws;
    float*          h    = (float*)(ws);                       // 51,200,000 B
    unsigned*       deg  = (unsigned*)(ws + 51200000);         //    200,000 B
    float*          dinv = (float*)(ws + 51400000);            //    200,000 B
    unsigned short* Whi  = (unsigned short*)(ws + 51600000);   //    458,752 B (16B aligned)
    unsigned short* Wlo  = (unsigned short*)(ws + 52058752);   //    458,752 B (16B aligned)

    // 1. W -> bf16 hi/lo fragments
    prep_w_frags<<<(NT_TOTAL * KT_TOTAL * 32 * 16 + 255) / 256, 256, 0, stream>>>(W, Whi, Wlo);

    // 2. H = X @ W via bf16x3 WMMA
    gemm_xw<<<(NN + BM - 1) / BM, 256, 0, stream>>>(span, types, emb, Whi, Wlo, h);

    // 3. degrees + dinv
    deg_init <<<(NN + 255) / 256, 256, 0, stream>>>(deg);
    deg_count<<<(NE + 255) / 256, 256, 0, stream>>>(ei, deg);
    calc_dinv<<<(NN + 255) / 256, 256, 0, stream>>>(deg, dinv);

    // 4. self-loop init of out, then edge scatter
    self_loop_init<<<(NN * HID + 255) / 256, 256, 0, stream>>>(h, dinv, out);
    scatter_edges <<<NE / 8, 256, 0, stream>>>(ei, h, dinv, out);

    // 5. bias + PReLU
    finalize<<<(NN * HID + 255) / 256, 256, 0, stream>>>(b, alpha, out);
}
